// Trans_89283780149720
// MI455X (gfx1250) — matrix operand order repaired and checked by reading, not compiled, not run on hardware
//
#include <hip/hip_runtime.h>
#include <math.h>

// Problem sizes (fixed by the reference)
#define BB   4
#define SS   1024
#define DDIM 1024
#define HH   16
#define DKK  64
#define DFFF 4096
#define ROWS (BB * SS)   // 4096

typedef __attribute__((ext_vector_type(16))) __bf16 v16bf;
typedef __attribute__((ext_vector_type(8)))  __bf16 v8bf;
typedef __attribute__((ext_vector_type(8)))  float  v8f;

static __device__ __forceinline__ v16bf cat8(v8bf lo, v8bf hi) {
    return __builtin_shufflevector(lo, hi, 0,1,2,3,4,5,6,7,8,9,10,11,12,13,14,15);
}

static __device__ __forceinline__ v8f wmma_bf16(v16bf a, v16bf b, v8f c) {
    // D = A(16x32 bf16) * B(32x16 bf16) + C(16x16 f32)
    return __builtin_amdgcn_wmma_f32_16x16x32_bf16(false, a, false, b, (short)0, c,
                                                   false, false);
}

// Async memory -> LDS copy of 16 bytes (gfx1250 GLOBAL_LOAD_ASYNC_TO_LDS_B128,
// GV mode, tracked by ASYNCcnt).  lds_addr: LDS byte offset (flat LDS address
// low 32 bits == LDS offset per aperture mapping), gptr: 64-bit global address.
static __device__ __forceinline__ void async_copy_b128(void* lds_ptr, const void* gptr) {
    unsigned lds_addr = (unsigned)(unsigned long long)lds_ptr;
    asm volatile("global_load_async_to_lds_b128 %0, %1, off"
                 :
                 : "v"(lds_addr), "v"(gptr)
                 : "memory");
}

static __device__ __forceinline__ void wait_asynccnt0() {
    asm volatile("s_wait_asynccnt 0" ::: "memory");
}

// ---------------------------------------------------------------------------
// f32 -> bf16 elementwise convert
// ---------------------------------------------------------------------------
__global__ void cvt_bf16_kernel(const float* __restrict__ in,
                                __bf16* __restrict__ out, int n) {
    int i = blockIdx.x * blockDim.x + threadIdx.x;
    if (i < n) out[i] = (__bf16)in[i];
}

// ---------------------------------------------------------------------------
// LayerNorm (torch-style: unbiased std, g*(x-mean)/(std+eps)+b), out bf16
// one 256-thread block per row of D=1024
// ---------------------------------------------------------------------------
__global__ __launch_bounds__(256)
void ln_kernel(const float* __restrict__ x, const float* __restrict__ g,
               const float* __restrict__ bta, __bf16* __restrict__ out) {
    const int row = blockIdx.x;
    const float* xr = x + (size_t)row * DDIM;
    __shared__ float red[256];

    float s1 = 0.f, s2 = 0.f;
    for (int i = threadIdx.x; i < DDIM; i += 256) {
        float v = xr[i];
        s1 += v;
        s2 += v * v;
    }
    red[threadIdx.x] = s1;
    __syncthreads();
    for (int off = 128; off > 0; off >>= 1) {
        if (threadIdx.x < (unsigned)off) red[threadIdx.x] += red[threadIdx.x + off];
        __syncthreads();
    }
    float sum1 = red[0];
    __syncthreads();
    red[threadIdx.x] = s2;
    __syncthreads();
    for (int off = 128; off > 0; off >>= 1) {
        if (threadIdx.x < (unsigned)off) red[threadIdx.x] += red[threadIdx.x + off];
        __syncthreads();
    }
    float sum2 = red[0];

    float mean = sum1 * (1.0f / DDIM);
    float var  = (sum2 - (float)DDIM * mean * mean) * (1.0f / (DDIM - 1));
    var = fmaxf(var, 0.0f);
    float inv = 1.0f / (sqrtf(var) + 1e-6f);

    __bf16* o = out + (size_t)row * DDIM;
    for (int i = threadIdx.x; i < DDIM; i += 256) {
        o[i] = (__bf16)(g[i] * (xr[i] - mean) * inv + bta[i]);
    }
}

// ---------------------------------------------------------------------------
// Async stage of one 128x64 bf16 tile pair (A rows bm.., W rows bn..) into LDS
// ---------------------------------------------------------------------------
static __device__ __forceinline__ void stage_tiles(const __bf16* __restrict__ A,
                                                   const __bf16* __restrict__ W,
                                                   __bf16* As, __bf16* Ws,
                                                   int tid, int bm, int bn,
                                                   int K, int k0) {
#pragma unroll
    for (int i = 0; i < 4; i++) {
        int c  = tid + 256 * i;          // 0..1023 chunks of 8 bf16
        int r  = c >> 3;                 // 8 chunks per row
        int kc = (c & 7) * 8;
        async_copy_b128(As + r * 64 + kc, A + (size_t)(bm + r) * K + k0 + kc);
        async_copy_b128(Ws + r * 64 + kc, W + (size_t)(bn + r) * K + k0 + kc);
    }
}

// ---------------------------------------------------------------------------
// Tiled WMMA GEMM:  C[M,N] = A[M,K] (bf16) * W[N,K]^T (bf16) + bias[N]
// block tile 128x128, 8 waves (4x2), wave tile 32x64 (2x4 WMMA frags).
// K tile = 64 (two WMMA k-steps), double-buffered async-to-LDS staging:
// tile t+1 streams in (ASYNCcnt) while tile t runs on the WMMA pipe.
// Epilogues: 0 -> bf16 store; 1 -> f32 store + residual; 2 -> SiLU -> bf16
// ---------------------------------------------------------------------------
template <int EPI>
__global__ __launch_bounds__(256)
void gemm_kernel(const __bf16* __restrict__ A, const __bf16* __restrict__ W,
                 const float* __restrict__ bias, const float* __restrict__ resid,
                 __bf16* __restrict__ outb, float* __restrict__ outf,
                 int M, int N, int K) {
    __shared__ alignas(32) __bf16 As[2][128 * 64];   // 2 x 16 KB
    __shared__ alignas(32) __bf16 Ws[2][128 * 64];   // 2 x 16 KB

    const int tid  = threadIdx.x;
    const int lane = tid & 31;
    const int wid  = tid >> 5;         // 0..7
    const int wm   = wid >> 1;         // 0..3
    const int wn   = wid & 1;          // 0..1
    const int m0   = 32 * wm;
    const int n0   = 64 * wn;
    const int mr   = lane & 15;
    const int hi   = lane >> 4;
    const int bm   = blockIdx.y * 128;
    const int bn   = blockIdx.x * 128;

    const v8f vzero = {0.f, 0.f, 0.f, 0.f, 0.f, 0.f, 0.f, 0.f};
    v8f acc[2][4];
#pragma unroll
    for (int i = 0; i < 2; i++)
#pragma unroll
        for (int j = 0; j < 4; j++) acc[i][j] = vzero;

    const int nk = K / 64;
    stage_tiles(A, W, As[0], Ws[0], tid, bm, bn, K, 0);

    for (int kt = 0; kt < nk; kt++) {
        const int cur = kt & 1;
        wait_asynccnt0();
        __syncthreads();   // also guarantees buffer cur^1 is no longer being read
        if (kt + 1 < nk)
            stage_tiles(A, W, As[cur ^ 1], Ws[cur ^ 1], tid, bm, bn, K,
                        (kt + 1) * 64);

        v16bf afr[2][2];   // [ks][fm]
        v16bf bfr[2][4];   // [ks][fn]
#pragma unroll
        for (int ks = 0; ks < 2; ks++) {
#pragma unroll
            for (int fm = 0; fm < 2; fm++) {
                const __bf16* p = As[cur] + (m0 + 16 * fm + mr) * 64 + 32 * ks;
                afr[ks][fm] = *(const v16bf*)(p + 16 * hi);
            }
#pragma unroll
            for (int fn = 0; fn < 4; fn++) {
                const __bf16* p = Ws[cur] + (n0 + 16 * fn + mr) * 64 + 32 * ks + 16 * hi;
                bfr[ks][fn] = *(const v16bf*)p;
            }
        }
#pragma unroll
        for (int ks = 0; ks < 2; ks++)
#pragma unroll
            for (int fm = 0; fm < 2; fm++)
#pragma unroll
                for (int fn = 0; fn < 4; fn++)
                    acc[fm][fn] = wmma_bf16(afr[ks][fm], bfr[ks][fn], acc[fm][fn]);
    }

    // epilogue: C row = bm+m0+16*fm+8*hi+v, col = bn+n0+16*fn+mr
#pragma unroll
    for (int fm = 0; fm < 2; fm++) {
#pragma unroll
        for (int fn = 0; fn < 4; fn++) {
            int col  = bn + n0 + 16 * fn + mr;
            float bv = bias[col];
#pragma unroll
            for (int v = 0; v < 8; v++) {
                int row = bm + m0 + 16 * fm + 8 * hi + v;
                size_t idx = (size_t)row * N + col;
                float xv = acc[fm][fn][v] + bv;
                if (EPI == 0) {
                    outb[idx] = (__bf16)xv;
                } else if (EPI == 1) {
                    outf[idx] = xv + resid[idx];
                } else {
                    float sv = xv / (1.0f + __expf(-xv));   // SiLU
                    outb[idx] = (__bf16)sv;
                }
            }
        }
    }
}

// ---------------------------------------------------------------------------
// Two-pass WMMA attention for one (b, h, 64-query tile).
// 128 threads = 4 waves; wave w owns query rows [16w, 16w+16).
// Pass 1: row max of masked, scaled scores. Pass 2: p=exp(s-max), sum, P@V.
// K tiles staged with async-to-LDS; Q fragments hoisted (loop-invariant).
// ---------------------------------------------------------------------------
__global__ __launch_bounds__(128)
void attn_kernel(const __bf16* __restrict__ q, const __bf16* __restrict__ k,
                 const __bf16* __restrict__ v, const int* __restrict__ mask,
                 __bf16* __restrict__ out) {
    const int qt = blockIdx.x;   // 0..15 (query tile)
    const int h  = blockIdx.y;   // 0..15
    const int b  = blockIdx.z;   // 0..3
    const int q0 = qt * 64;

    const int tid  = threadIdx.x;
    const int lane = tid & 31;
    const int wid  = tid >> 5;   // 0..3
    const int mr   = lane & 15;
    const int hi   = lane >> 4;

    __shared__ alignas(32) __bf16 Qs[64 * 64];       // [qr][d]
    __shared__ alignas(32) __bf16 Ks[64 * 64];       // [kr][d]
    __shared__ alignas(32) __bf16 Vt[64 * 64];       // [d][kr] (transposed)
    __shared__ alignas(32) __bf16 Ps[4][16 * 64];    // per-wave P tile [qr][kr]

    const size_t hoff = (size_t)h * DKK;
    const v8f vzero = {0.f, 0.f, 0.f, 0.f, 0.f, 0.f, 0.f, 0.f};

    // load Q tile (64x64 bf16) via async copy
#pragma unroll
    for (int i = 0; i < 4; i++) {
        int c = tid + 128 * i;      // 0..511 chunks of 8
        int r = c >> 3, dc = (c & 7) * 8;
        async_copy_b128(Qs + r * 64 + dc,
                        q + (size_t)(b * SS + q0 + r) * DDIM + hoff + dc);
    }
    wait_asynccnt0();
    __syncthreads();

    // Q A-fragments: invariant across all k-tiles and both passes
    v16bf aq[2];
#pragma unroll
    for (int ks = 0; ks < 2; ks++) {
        const __bf16* p = Qs + (16 * wid + mr) * 64 + 32 * ks;
        aq[ks] = *(const v16bf*)(p + 16 * hi);
    }

    float rowmax[8], rowsum[8];
#pragma unroll
    for (int e = 0; e < 8; e++) { rowmax[e] = -1e30f; rowsum[e] = 0.f; }

    // ---------------- pass 1: row max ----------------
    for (int jt = 0; jt < 16; jt++) {
#pragma unroll
        for (int i = 0; i < 4; i++) {
            int c = tid + 128 * i;
            int r = c >> 3, dc = (c & 7) * 8;
            async_copy_b128(Ks + r * 64 + dc,
                            k + (size_t)(b * SS + jt * 64 + r) * DDIM + hoff + dc);
        }
        wait_asynccnt0();
        __syncthreads();

#pragma unroll
        for (int fn = 0; fn < 4; fn++) {
            v8f s = vzero;
#pragma unroll
            for (int ks = 0; ks < 2; ks++) {
                const __bf16* p = Ks + (16 * fn + mr) * 64 + 32 * ks + 16 * hi;
                s = wmma_bf16(aq[ks], *(const v16bf*)p, s);
            }
            int col = jt * 64 + 16 * fn + mr;
            int mv  = mask[b * SS + col];
#pragma unroll
            for (int e = 0; e < 8; e++) {
                float sv = (mv == 0) ? -1e9f : s[e] * 0.125f;  // 1/sqrt(64)
                rowmax[e] = fmaxf(rowmax[e], sv);
            }
        }
        __syncthreads();
    }
    // reduce max across the 16-lane group holding each row
#pragma unroll
    for (int e = 0; e < 8; e++) {
#pragma unroll
        for (int off = 1; off < 16; off <<= 1)
            rowmax[e] = fmaxf(rowmax[e], __shfl_xor(rowmax[e], off, 32));
    }

    v8f o[4];
#pragma unroll
    for (int dn = 0; dn < 4; dn++) o[dn] = vzero;

    // ---------------- pass 2: exp + P@V ----------------
    for (int jt = 0; jt < 16; jt++) {
#pragma unroll
        for (int i = 0; i < 4; i++) {
            int c = tid + 128 * i;
            int r = c >> 3, dc = (c & 7) * 8;
            async_copy_b128(Ks + r * 64 + dc,
                            k + (size_t)(b * SS + jt * 64 + r) * DDIM + hoff + dc);
        }
        // V tile, stored transposed [d][kr]
#pragma unroll
        for (int i = 0; i < 32; i++) {
            int e  = tid + 128 * i;     // 0..4095
            int kr = e >> 6, d = e & 63;
            Vt[d * 64 + kr] = v[(size_t)(b * SS + jt * 64 + kr) * DDIM + hoff + d];
        }
        wait_asynccnt0();
        __syncthreads();

#pragma unroll
        for (int fn = 0; fn < 4; fn++) {
            v8f s = vzero;
#pragma unroll
            for (int ks = 0; ks < 2; ks++) {
                const __bf16* p = Ks + (16 * fn + mr) * 64 + 32 * ks + 16 * hi;
                s = wmma_bf16(aq[ks], *(const v16bf*)p, s);
            }
            int col = jt * 64 + 16 * fn + mr;
            int mv  = mask[b * SS + col];
#pragma unroll
            for (int e = 0; e < 8; e++) {
                float sv = (mv == 0) ? -1e9f : s[e] * 0.125f;
                float pe = __expf(sv - rowmax[e]);
                rowsum[e] += pe;
                Ps[wid][(8 * hi + e) * 64 + 16 * fn + mr] = (__bf16)pe;
            }
        }
        __syncthreads();

        // o += P @ V   (A = Ps[wid] 16x64, B = Vt columns)
#pragma unroll
        for (int ks = 0; ks < 2; ks++) {
            const __bf16* pa = Ps[wid] + mr * 64 + 32 * ks;
            v16bf ap = *(const v16bf*)(pa + 16 * hi);
#pragma unroll
            for (int dn = 0; dn < 4; dn++) {
                const __bf16* pb = Vt + (16 * dn + mr) * 64 + 32 * ks + 16 * hi;
                o[dn] = wmma_bf16(ap, *(const v16bf*)pb, o[dn]);
            }
        }
        __syncthreads();
    }

    // reduce row sums across the 16-lane group, normalize, store bf16
#pragma unroll
    for (int e = 0; e < 8; e++) {
#pragma unroll
        for (int off = 1; off < 16; off <<= 1)
            rowsum[e] += __shfl_xor(rowsum[e], off, 32);
    }
#pragma unroll
    for (int dn = 0; dn < 4; dn++) {
#pragma unroll
        for (int e = 0; e < 8; e++) {
            float val = o[dn][e] / rowsum[e];
            int qr = q0 + 16 * wid + 8 * hi + e;
            out[(size_t)(b * SS + qr) * DDIM + hoff + 16 * dn + mr] = (__bf16)val;
        }
    }
}

// ---------------------------------------------------------------------------
// Host-side orchestration
// ---------------------------------------------------------------------------
extern "C" void kernel_launch(void* const* d_in, const int* in_sizes, int n_in,
                              void* d_out, int out_size, void* d_ws, size_t ws_size,
                              hipStream_t stream) {
    (void)in_sizes; (void)n_in; (void)out_size; (void)ws_size;

    const float* y    = (const float*)d_in[0];
    const float* z    = (const float*)d_in[1];
    const int*   mask = (const int*)d_in[2];
    const float* sa_w = (const float*)d_in[3];
    const float* sa_b = (const float*)d_in[4];
    const float* ca_w = (const float*)d_in[5];
    const float* ca_b = (const float*)d_in[6];
    const float* w1   = (const float*)d_in[7];
    const float* b1   = (const float*)d_in[8];
    const float* w2   = (const float*)d_in[9];
    const float* b2   = (const float*)d_in[10];
    const float* ln_g = (const float*)d_in[11];
    const float* ln_b = (const float*)d_in[12];

    char* ws  = (char*)d_ws;
    size_t off = 0;
    auto carve = [&](size_t bytes) -> char* {
        char* p = ws + off;
        off += (bytes + 255) & ~(size_t)255;
        return p;
    };

    const size_t DD2 = (size_t)DDIM * DDIM;
    __bf16* wsa   = (__bf16*)carve(4 * DD2 * sizeof(__bf16));        // self-attn weights
    __bf16* wca   = (__bf16*)carve(4 * DD2 * sizeof(__bf16));        // cross-attn weights
    __bf16* ww1   = (__bf16*)carve((size_t)DFFF * DDIM * sizeof(__bf16));
    __bf16* ww2   = (__bf16*)carve((size_t)DDIM * DFFF * sizeof(__bf16));
    __bf16* zb    = (__bf16*)carve((size_t)ROWS * DDIM * sizeof(__bf16));
    __bf16* yn    = (__bf16*)carve((size_t)ROWS * DDIM * sizeof(__bf16));
    __bf16* qb    = (__bf16*)carve((size_t)ROWS * DDIM * sizeof(__bf16));
    __bf16* kb    = (__bf16*)carve((size_t)ROWS * DDIM * sizeof(__bf16));
    __bf16* vb    = (__bf16*)carve((size_t)ROWS * DDIM * sizeof(__bf16));
    __bf16* attnb = (__bf16*)carve((size_t)ROWS * DDIM * sizeof(__bf16));
    __bf16* hb    = (__bf16*)carve((size_t)ROWS * DFFF * sizeof(__bf16));
    float*  y1    = (float*)carve((size_t)ROWS * DDIM * sizeof(float));
    float*  y2    = (float*)carve((size_t)ROWS * DDIM * sizeof(float));

    auto cvt = [&](const float* src, __bf16* dst, int n) {
        cvt_bf16_kernel<<<(n + 255) / 256, 256, 0, stream>>>(src, dst, n);
    };

    // weight / z converts
    cvt(sa_w, wsa, (int)(4 * DD2));
    cvt(ca_w, wca, (int)(4 * DD2));
    cvt(w1, ww1, DFFF * DDIM);
    cvt(w2, ww2, DDIM * DFFF);
    cvt(z, zb, ROWS * DDIM);

    dim3 gProj(DDIM / 128, ROWS / 128);   // N=1024
    dim3 gFfn1(DFFF / 128, ROWS / 128);   // N=4096
    dim3 gAttn(SS / 64, HH, BB);

    // ---- sublayer 0: self-attention ----
    ln_kernel<<<ROWS, 256, 0, stream>>>(y, ln_g + 0 * DDIM, ln_b + 0 * DDIM, yn);
    gemm_kernel<0><<<gProj, 256, 0, stream>>>(yn, wsa + 0 * DD2, sa_b + 0 * DDIM,
                                              nullptr, qb, nullptr, ROWS, DDIM, DDIM);
    gemm_kernel<0><<<gProj, 256, 0, stream>>>(yn, wsa + 1 * DD2, sa_b + 1 * DDIM,
                                              nullptr, kb, nullptr, ROWS, DDIM, DDIM);
    gemm_kernel<0><<<gProj, 256, 0, stream>>>(yn, wsa + 2 * DD2, sa_b + 2 * DDIM,
                                              nullptr, vb, nullptr, ROWS, DDIM, DDIM);
    attn_kernel<<<gAttn, 128, 0, stream>>>(qb, kb, vb, mask, attnb);
    gemm_kernel<1><<<gProj, 256, 0, stream>>>(attnb, wsa + 3 * DD2, sa_b + 3 * DDIM,
                                              y, nullptr, y1, ROWS, DDIM, DDIM);

    // ---- sublayer 1: cross-attention (k/v from raw z) ----
    ln_kernel<<<ROWS, 256, 0, stream>>>(y1, ln_g + 1 * DDIM, ln_b + 1 * DDIM, yn);
    gemm_kernel<0><<<gProj, 256, 0, stream>>>(yn, wca + 0 * DD2, ca_b + 0 * DDIM,
                                              nullptr, qb, nullptr, ROWS, DDIM, DDIM);
    gemm_kernel<0><<<gProj, 256, 0, stream>>>(zb, wca + 1 * DD2, ca_b + 1 * DDIM,
                                              nullptr, kb, nullptr, ROWS, DDIM, DDIM);
    gemm_kernel<0><<<gProj, 256, 0, stream>>>(zb, wca + 2 * DD2, ca_b + 2 * DDIM,
                                              nullptr, vb, nullptr, ROWS, DDIM, DDIM);
    attn_kernel<<<gAttn, 128, 0, stream>>>(qb, kb, vb, mask, attnb);
    gemm_kernel<1><<<gProj, 256, 0, stream>>>(attnb, wca + 3 * DD2, ca_b + 3 * DDIM,
                                              y1, nullptr, y2, ROWS, DDIM, DDIM);

    // ---- sublayer 2: FFN with SiLU ----
    ln_kernel<<<ROWS, 256, 0, stream>>>(y2, ln_g + 2 * DDIM, ln_b + 2 * DDIM, yn);
    gemm_kernel<2><<<gFfn1, 256, 0, stream>>>(yn, ww1, b1, nullptr, hb, nullptr,
                                              ROWS, DFFF, DDIM);
    gemm_kernel<1><<<gProj, 256, 0, stream>>>(hb, ww2, b2, y2, nullptr,
                                              (float*)d_out, ROWS, DDIM, DFFF);
}
